// DacaDecoderLayer_2585570312448
// MI455X (gfx1250) — matrix operand order repaired and checked by reading, not compiled, not run on hardware
//
#include <hip/hip_runtime.h>
#include <hip/hip_bf16.h>

// ---------------------------------------------------------------------------
// DacaDecoderLayer for MI455X (gfx1250): f16 WMMA GEMM pipeline, f32 epilogues.
// Async global->LDS staging (ASYNCcnt) + double-buffered software pipeline.
// B=8, T=S=512, D=512, H=8, DH=64.
// ---------------------------------------------------------------------------

typedef __attribute__((ext_vector_type(16))) _Float16 v16h;
typedef __attribute__((ext_vector_type(8)))  _Float16 v8h;
typedef __attribute__((ext_vector_type(8)))  float    v8f;

#define CB 8
#define CT 512
#define CS 512
#define CD 512
#define CH 8
#define CDH 64
#define ATT_SCALE 0.125f   // 1/sqrt(64)

// ---- CDNA5 async global->LDS copy (16B per lane), tracked by ASYNCcnt ----
__device__ __forceinline__ void async_b128(unsigned int lds_off, const void* g)
{
    asm volatile("global_load_async_to_lds_b128 %0, %1, off"
                 :: "v"(lds_off), "v"((unsigned long long)(size_t)g)
                 : "memory");
}
__device__ __forceinline__ void wait_async0()
{
#if __has_builtin(__builtin_amdgcn_s_wait_asynccnt)
    __builtin_amdgcn_s_wait_asynccnt(0);
#else
    asm volatile("s_wait_asynccnt 0" ::: "memory");
#endif
}

// ===========================================================================
// WMMA GEMM: C[m,n] = act( sum_k A[m,k] * B[k,n] + bias[n] )
// A: f16 (M,K) row-major, lda. B: f16 (K,N) row-major, ldb. C: f32, ldc.
// Batched over blockIdx.z with two-level strides (z -> b1 = z/nb2, b2 = z%nb2).
// Block: 256 thr (8 waves). Tile: 128(M) x BNT(N) x 32(K), double-buffered LDS,
// async-DMA staging. Wave: 32 x BNT/2 (2 x BNT/32 frags of 16x16x32 WMMA).
// ===========================================================================
template <int BNT>
__global__ __launch_bounds__(256)
void gemm_wmma_kernel(const _Float16* __restrict__ A, const _Float16* __restrict__ B,
                      const float* __restrict__ bias, float* __restrict__ C,
                      int M, int N, int K, int lda, int ldb, int ldc,
                      long asb1, long asb2, long bsb1, long bsb2, long csb1, long csb2,
                      int nb2, int relu)
{
    constexpr int FN = BNT / 32;                   // N frags per wave
    __shared__ _Float16 As[2][128][48];            // 32 K halves + pad (96B rows)
    __shared__ _Float16 Bs[2][32][BNT + 8];        // BNT N halves + pad (16B-mult rows)

    const int z  = blockIdx.z;
    const int b1 = z / nb2, b2 = z % nb2;
    A += (long)b1 * asb1 + (long)b2 * asb2;
    B += (long)b1 * bsb1 + (long)b2 * bsb2;
    C += (long)b1 * csb1 + (long)b2 * csb2;

    const int tid  = threadIdx.x;
    const int lane = tid & 31;
    const int wave = tid >> 5;
    const int mw   = (wave >> 1) * 32;             // wave M offset: 0,32,64,96
    const int nw   = (wave & 1) * (BNT / 2);       // wave N offset

    const int m0 = blockIdx.y * 128;
    const int n0 = blockIdx.x * BNT;

    v8f acc[2][FN] = {};

    // staging indices (per thread)
    const int arow = tid >> 2;                     // 0..63 (two row passes)
    const int acol = (tid & 3) * 8;                // halves
    const int brow = tid >> 3;                     // 0..31 (K)
    const int bcol = (tid & 7) * 8;                // halves

    // fragment lane decomposition
    const int fr  = lane & 15;                     // A row / B,C N index
    const int hi  = lane >> 4;
    const int akb = hi * 8;

    const _Float16* Ag0 = &A[(long)(m0 + arow) * lda + acol];
    const _Float16* Ag1 = &A[(long)(m0 + arow + 64) * lda + acol];
    const _Float16* Bg  = &B[(long)brow * ldb + n0 + bcol];

    const int nk = K >> 5;

    auto issue_tile = [&](int kt) {
        const int buf = kt & 1;
        const long ko = (long)kt * 32;
        async_b128((unsigned int)(size_t)&As[buf][arow][acol],      Ag0 + ko);
        async_b128((unsigned int)(size_t)&As[buf][arow + 64][acol], Ag1 + ko);
        #pragma unroll
        for (int i = 0; i < BNT / 64; ++i)
            async_b128((unsigned int)(size_t)&Bs[buf][brow][bcol + i * 64],
                       Bg + ko * ldb + i * 64);
    };

    issue_tile(0);

    for (int kt = 0; kt < nk; ++kt) {
        const int buf = kt & 1;
        wait_async0();              // own tile-kt DMA done
        __syncthreads();            // everyone's DMA done; prev-buffer reads done
        if (kt + 1 < nk) issue_tile(kt + 1);   // overlap next DMA with math

        v16h afrag[2], bfrag[FN];
        #pragma unroll
        for (int fm = 0; fm < 2; ++fm) {
            const _Float16* ap = &As[buf][mw + fm * 16 + fr][akb];
            v8h lo = *(const v8h*)(ap);
            v8h hh = *(const v8h*)(ap + 16);
            #pragma unroll
            for (int i = 0; i < 8; ++i) { afrag[fm][i] = lo[i]; afrag[fm][8 + i] = hh[i]; }
        }
        #pragma unroll
        for (int fn = 0; fn < FN; ++fn) {
            const int bn = nw + fn * 16 + fr;
            #pragma unroll
            for (int i = 0; i < 8; ++i) { bfrag[fn][i] = Bs[buf][akb + i][bn]; bfrag[fn][8 + i] = Bs[buf][16 + akb + i][bn]; }
        }
        #pragma unroll
        for (int fm = 0; fm < 2; ++fm)
            #pragma unroll
            for (int fn = 0; fn < FN; ++fn)
                acc[fm][fn] = __builtin_amdgcn_wmma_f32_16x16x32_f16(
                    false, afrag[fm], false, bfrag[fn], (short)0, acc[fm][fn], false, false);
    }

    // epilogue: C frag layout -> lane fr = N, vgpr r + 8*hi = M
    #pragma unroll
    for (int fm = 0; fm < 2; ++fm) {
        #pragma unroll
        for (int fn = 0; fn < FN; ++fn) {
            const int gn = n0 + nw + fn * 16 + fr;
            const float bv = bias ? bias[gn] : 0.0f;
            #pragma unroll
            for (int r = 0; r < 8; ++r) {
                const int gm = m0 + mw + fm * 16 + hi * 8 + r;
                float val = acc[fm][fn][r] + bv;
                if (relu) val = fmaxf(val, 0.0f);
                C[(long)gm * ldc + gn] = val;
            }
        }
    }
}

// ===========================================================================
// f32 -> f16 flat convert with scale
// ===========================================================================
__global__ void convert_f16_kernel(const float* __restrict__ src, _Float16* __restrict__ dst,
                                   long n, float scale)
{
    long i = (long)blockIdx.x * blockDim.x + threadIdx.x;
    if (i < n) dst[i] = (_Float16)(src[i] * scale);
}

// ===========================================================================
// batched f32 (R,C) -> f16 (C,R) transpose-convert. dst ld = R.
// ===========================================================================
__global__ __launch_bounds__(256)
void transpose_convert_kernel(const float* __restrict__ src, _Float16* __restrict__ dst,
                              int R, int C, int srs,
                              long sb1, long sb2, int nb2, long dstride, float scale)
{
    const int z = blockIdx.z;
    src += (long)(z / nb2) * sb1 + (long)(z % nb2) * sb2;
    dst += (long)z * dstride;
    __shared__ float tile[32][33];
    const int c0 = blockIdx.x * 32, r0 = blockIdx.y * 32;
    const int tx = threadIdx.x & 31, ty = threadIdx.x >> 5;   // 32 x 8
    #pragma unroll
    for (int i = 0; i < 32; i += 8)
        tile[ty + i][tx] = src[(long)(r0 + ty + i) * srs + (c0 + tx)];
    __syncthreads();
    #pragma unroll
    for (int i = 0; i < 32; i += 8)
        dst[(long)(c0 + ty + i) * R + (r0 + tx)] = (_Float16)(tile[tx][ty + i] * scale);
}

// ===========================================================================
// qg[b,h,t,d] = qW[b,h,t,d] * gate[t,d]  (f32 -> f16)
// ===========================================================================
__global__ void gatemul_kernel(const float* __restrict__ qW, const float* __restrict__ gate,
                               _Float16* __restrict__ out, long total)
{
    long i = (long)blockIdx.x * blockDim.x + threadIdx.x;
    if (i >= total) return;
    const int d = (int)(i % CD);
    const int t = (int)((i / CD) % CT);
    out[i] = (_Float16)(qW[i] * gate[(long)t * CD + d]);
}

// ===========================================================================
// cbias[b,t,h] = scale * sum_e cq[b,t,h*DH+e] * bk[h*DH+e]
// ===========================================================================
__global__ void cbias_kernel(const float* __restrict__ cq, const float* __restrict__ bk,
                             float* __restrict__ cbias, int total, float scale)
{
    int i = blockIdx.x * blockDim.x + threadIdx.x;   // over B*T*H
    if (i >= total) return;
    const int h = i % CH;
    const long bt = i / CH;
    const float* q = cq + bt * (long)CD + h * CDH;
    const float* w = bk + h * CDH;
    float s = 0.f;
    #pragma unroll 8
    for (int e = 0; e < CDH; ++e) s += q[e] * w[e];
    cbias[i] = s * scale;
}

// ===========================================================================
// Row softmax over S=512, wave per row. self: +mask[t,s]; cross: +cbias[b,t,h].
// Writes f16 attn.
// ===========================================================================
__global__ __launch_bounds__(256)
void softmax_kernel(const float* __restrict__ S, const float* __restrict__ mask,
                    const float* __restrict__ cbias, _Float16* __restrict__ out, int Sn)
{
    const int wid = threadIdx.x >> 5, lane = threadIdx.x & 31;
    const long row = (long)blockIdx.x * 8 + wid;     // over B*H*T
    const int t = (int)(row % CT);
    const long bh = row / CT;
    const int h = (int)(bh % CH);
    const long b = bh / CH;

    float add = 0.f;
    const float* mrow = nullptr;
    if (mask)  mrow = mask + (long)t * Sn;
    if (cbias) add = cbias[(b * CT + t) * CH + h];

    const float* sr = S + row * Sn;
    const int cnt = Sn >> 5;                         // 16
    float v[16];
    float mx = -3.0e38f;
    for (int j = 0; j < cnt; ++j) {
        float x = sr[lane + j * 32] + (mrow ? mrow[lane + j * 32] : add);
        v[j] = x; mx = fmaxf(mx, x);
    }
    for (int o = 16; o; o >>= 1) mx = fmaxf(mx, __shfl_xor(mx, o, 32));
    float sum = 0.f;
    for (int j = 0; j < cnt; ++j) { v[j] = __expf(v[j] - mx); sum += v[j]; }
    for (int o = 16; o; o >>= 1) sum += __shfl_xor(sum, o, 32);
    const float inv = 1.0f / sum;
    _Float16* orow = out + row * Sn;
    for (int j = 0; j < cnt; ++j) orow[lane + j * 32] = (_Float16)(v[j] * inv);
}

// ===========================================================================
// y = LN(res + x) * g + b ; wave per row of D=512. Writes f32 (+ optional f16).
// ===========================================================================
__global__ __launch_bounds__(256)
void residual_ln_kernel(const float* __restrict__ res, const float* __restrict__ x,
                        const float* __restrict__ g, const float* __restrict__ bta,
                        float* __restrict__ outf, _Float16* __restrict__ outh)
{
    const int wid = threadIdx.x >> 5, lane = threadIdx.x & 31;
    const long row = (long)blockIdx.x * 8 + wid;
    const float* r = res + row * CD;
    const float* xx = x + row * CD;
    const int cnt = CD >> 5;                         // 16
    float v[16];
    float s = 0.f, s2 = 0.f;
    for (int j = 0; j < cnt; ++j) {
        float y = r[lane + j * 32] + xx[lane + j * 32];
        v[j] = y; s += y; s2 += y * y;
    }
    for (int o = 16; o; o >>= 1) { s += __shfl_xor(s, o, 32); s2 += __shfl_xor(s2, o, 32); }
    const float mean = s * (1.0f / CD);
    const float var  = s2 * (1.0f / CD) - mean * mean;
    const float rstd = rsqrtf(var + 1e-5f);
    for (int j = 0; j < cnt; ++j) {
        const int c = lane + j * 32;
        const float y = (v[j] - mean) * rstd * g[c] + bta[c];
        outf[row * CD + c] = y;
        if (outh) outh[row * CD + c] = (_Float16)y;
    }
}

// ===========================================================================
// Host side
// ===========================================================================
static inline size_t aln(size_t x) { return (x + 255) & ~(size_t)255; }

template <int BNT>
static void gemm_t(hipStream_t s, const _Float16* A, const _Float16* B, const float* bias,
                   float* C, int M, int N, int K, int lda, int ldb, int ldc,
                   long a1, long a2, long b1, long b2, long c1, long c2,
                   int nb2, int batches, int relu)
{
    dim3 g(N / BNT, M / 128, batches);
    gemm_wmma_kernel<BNT><<<g, 256, 0, s>>>(A, B, bias, C, M, N, K, lda, ldb, ldc,
                                            a1, a2, b1, b2, c1, c2, nb2, relu);
}

static void gemm(hipStream_t s, const _Float16* A, const _Float16* B, const float* bias,
                 float* C, int M, int N, int K, int lda, int ldb, int ldc,
                 long a1, long a2, long b1, long b2, long c1, long c2,
                 int nb2, int batches, int relu)
{
    if (N % 128 == 0)
        gemm_t<128>(s, A, B, bias, C, M, N, K, lda, ldb, ldc, a1, a2, b1, b2, c1, c2, nb2, batches, relu);
    else
        gemm_t<64>(s, A, B, bias, C, M, N, K, lda, ldb, ldc, a1, a2, b1, b2, c1, c2, nb2, batches, relu);
}

static void cvt(hipStream_t s, const float* src, _Float16* dst, long n, float scale)
{
    convert_f16_kernel<<<dim3((unsigned)((n + 255) / 256)), 256, 0, s>>>(src, dst, n, scale);
}

static void tconv(hipStream_t s, const float* src, _Float16* dst, int R, int C, int srs,
                  long sb1, long sb2, int nb2, long dstride, int batches, float scale)
{
    dim3 g(C / 32, R / 32, batches);
    transpose_convert_kernel<<<g, 256, 0, s>>>(src, dst, R, C, srs, sb1, sb2, nb2, dstride, scale);
}

extern "C" void kernel_launch(void* const* d_in, const int* in_sizes, int n_in,
                              void* d_out, int out_size, void* d_ws, size_t ws_size,
                              hipStream_t stream)
{
    const float* tgt      = (const float*)d_in[0];
    const float* memory   = (const float*)d_in[1];
    const float* tgt_mask = (const float*)d_in[2];
    const float* gate     = (const float*)d_in[3];
    const float* sa_in_w  = (const float*)d_in[4];
    const float* sa_in_b  = (const float*)d_in[5];
    const float* sa_out_w = (const float*)d_in[6];
    const float* sa_out_b = (const float*)d_in[7];
    const float* ca_in_w  = (const float*)d_in[8];
    const float* ca_in_b  = (const float*)d_in[9];
    const float* ca_out_w = (const float*)d_in[10];
    const float* ca_out_b = (const float*)d_in[11];
    const float* ff1_w    = (const float*)d_in[12];
    const float* ff1_b    = (const float*)d_in[13];
    const float* ff2_w    = (const float*)d_in[14];
    const float* ff2_b    = (const float*)d_in[15];
    const float* ln1_g    = (const float*)d_in[16];
    const float* ln1_b    = (const float*)d_in[17];
    const float* ln2_g    = (const float*)d_in[18];
    const float* ln2_b    = (const float*)d_in[19];
    const float* ln3_g    = (const float*)d_in[20];
    const float* ln3_b    = (const float*)d_in[21];
    float* outp = (float*)d_out;

    const long MROWS = (long)CB * CT;           // 4096
    const long NN    = MROWS * CD;              // 2M   (B,T,D)
    const long NSC   = (long)CB * CH * CT * CS; // 16.7M (B,H,T,S)
    const long NFF   = MROWS * 4 * CD;          // 8.4M

    // ---- workspace layout (bump allocator) ----
    char* p = (char*)d_ws;
    size_t off = 0;
    auto allocF = [&](long n) { float* r = (float*)(p + off); off = aln(off + (size_t)n * 4); return r; };
    auto allocH = [&](long n) { _Float16* r = (_Float16*)(p + off); off = aln(off + (size_t)n * 2); return r; };

    float* qf      = allocF(NN);      // also cq (f32) later
    float* kf      = allocF(NN);      // also cv (f32) later
    float* vf      = allocF(NN);      // also attn-context (f32) later
    float* projf   = allocF(NN);      // proj output before LN (reused 3x)
    float* x1f     = allocF(NN);
    float* x2f     = allocF(NN);
    float* cbiasf  = allocF((long)CB * CT * CH);
    float* scoresf = allocF(NSC);     // also qW (B,H,T,D)
    float* h1f     = allocF(NFF);

    _Float16* tgt_h  = allocH(NN);
    _Float16* mem_h  = allocH(NN);
    _Float16* qs_h   = allocH(NN);    // scaled q / scaled cq (reused)
    _Float16* kT_h   = allocH(NN);    // (B,H,DH,T)
    _Float16* v_h    = allocH(NN);    // v f16 / cv f16 (reused)
    _Float16* ctx_h  = allocH(NN);
    _Float16* x1h    = allocH(NN);
    _Float16* x2h    = allocH(NN);
    _Float16* memT_h = allocH(NN);    // (B,D,S)
    _Float16* qg_h   = allocH(NSC);   // (B,H,T,D)
    _Float16* attn_h = allocH(NSC);   // (B,H,T,S)
    _Float16* h1h    = allocH(NFF);

    _Float16* saT   = allocH((long)CD * 3 * CD);   // (512,1536) = sa_in_w^T
    _Float16* caT   = allocH((long)CD * 3 * CD);   // (512,1536) = ca_in_w^T
    _Float16* cwk_h = allocH((long)CD * CD);       // ca_in_w rows [512,1024) as-is (K,N)
    _Float16* soT   = allocH((long)CD * CD);
    _Float16* coT   = allocH((long)CD * CD);
    _Float16* f1T   = allocH((long)CD * 4 * CD);   // (512,2048)
    _Float16* f2T   = allocH((long)4 * CD * CD);   // (2048,512)
    (void)ws_size; (void)in_sizes; (void)n_in; (void)out_size;

    // ---- weight preparation ----
    tconv(stream, sa_in_w,  saT, 3 * CD, CD, CD, 0, 0, 1, 0, 1, 1.f);
    tconv(stream, ca_in_w,  caT, 3 * CD, CD, CD, 0, 0, 1, 0, 1, 1.f);
    tconv(stream, sa_out_w, soT, CD, CD, CD, 0, 0, 1, 0, 1, 1.f);
    tconv(stream, ca_out_w, coT, CD, CD, CD, 0, 0, 1, 0, 1, 1.f);
    tconv(stream, ff1_w,    f1T, 4 * CD, CD, CD, 0, 0, 1, 0, 1, 1.f);
    tconv(stream, ff2_w,    f2T, CD, 4 * CD, 4 * CD, 0, 0, 1, 0, 1, 1.f);
    cvt(stream, ca_in_w + (long)CD * CD, cwk_h, (long)CD * CD, 1.f);

    cvt(stream, tgt, tgt_h, NN, 1.f);
    cvt(stream, memory, mem_h, NN, 1.f);

    // ---- self attention ----
    // Q/K/V projections: (4096,512) x (512,512)
    gemm(stream, tgt_h, saT,        sa_in_b,        qf, (int)MROWS, CD, CD, CD, 3*CD, CD, 0,0,0,0,0,0, 1, 1, 0);
    gemm(stream, tgt_h, saT + CD,   sa_in_b + CD,   kf, (int)MROWS, CD, CD, CD, 3*CD, CD, 0,0,0,0,0,0, 1, 1, 0);
    gemm(stream, tgt_h, saT + 2*CD, sa_in_b + 2*CD, vf, (int)MROWS, CD, CD, CD, 3*CD, CD, 0,0,0,0,0,0, 1, 1, 0);

    cvt(stream, qf, qs_h, NN, ATT_SCALE);
    // K -> (B,H,DH,T) f16
    tconv(stream, kf, kT_h, CT, CDH, CD, (long)CT * CD, CDH, CH, (long)CDH * CT, CB * CH, 1.f);
    cvt(stream, vf, v_h, NN, 1.f);

    // scores[b,h,t,s] = q . k : per-(b,h) (512 x 512 x 64)
    gemm(stream, qs_h, kT_h, nullptr, scoresf, CT, CT, CDH, CD, CT, CT,
         (long)CT * CD, CDH, (long)CH * CDH * CT, (long)CDH * CT,
         (long)CH * CT * CT, (long)CT * CT, CH, CB * CH, 0);
    softmax_kernel<<<dim3((unsigned)(CB * CH * CT / 8)), 256, 0, stream>>>(scoresf, tgt_mask, nullptr, attn_h, CS);
    // ctx = attn @ V : per-(b,h) (512 x 64 x 512) -> (B,T,H,DH) in vf
    gemm(stream, attn_h, v_h, nullptr, vf, CT, CDH, CS, CS, CD, CD,
         (long)CH * CT * CS, (long)CT * CS, (long)CS * CD, CDH,
         (long)CT * CD, CDH, CH, CB * CH, 0);
    cvt(stream, vf, ctx_h, NN, 1.f);
    gemm(stream, ctx_h, soT, sa_out_b, projf, (int)MROWS, CD, CD, CD, CD, CD, 0,0,0,0,0,0, 1, 1, 0);
    residual_ln_kernel<<<dim3((unsigned)(MROWS / 8)), 256, 0, stream>>>(tgt, projf, ln1_g, ln1_b, x1f, x1h);

    // ---- gated cross attention ----
    // cq = x1 @ cwq^T + cbq   (into qf)
    gemm(stream, x1h, caT, ca_in_b, qf, (int)MROWS, CD, CD, CD, 3*CD, CD, 0,0,0,0,0,0, 1, 1, 0);
    cvt(stream, qf, qs_h, NN, ATT_SCALE);                     // scaled cq
    cbias_kernel<<<dim3((unsigned)((CB * CT * CH + 255) / 256)), 256, 0, stream>>>(
        qf, ca_in_b + CD, cbiasf, CB * CT * CH, ATT_SCALE);

    // qW[b,h,t,:] = cq_h @ cwk_h : per-(b,h) (512 x 512 x 64) -> scoresf as (B,H,T,D)
    gemm(stream, qs_h, cwk_h, nullptr, scoresf, CT, CD, CDH, CD, CD, CD,
         (long)CT * CD, CDH, 0, (long)CDH * CD,
         (long)CH * CT * CD, (long)CT * CD, CH, CB * CH, 0);
    gatemul_kernel<<<dim3((unsigned)((NSC + 255) / 256)), 256, 0, stream>>>(scoresf, gate, qg_h, NSC);

    // memory^T per b -> (B,D,S) f16
    tconv(stream, memory, memT_h, CS, CD, CD, (long)CS * CD, 0, 1, (long)CD * CS, CB, 1.f);
    // cv = memory @ cwv^T + cbv   (into kf)
    gemm(stream, mem_h, caT + 2*CD, ca_in_b + 2*CD, kf, (int)MROWS, CD, CD, CD, 3*CD, CD, 0,0,0,0,0,0, 1, 1, 0);
    cvt(stream, kf, v_h, NN, 1.f);                            // cv f16

    // cscores[b,h,t,s] = qg . memory : per-(b,h) (512 x 512 x 512)
    gemm(stream, qg_h, memT_h, nullptr, scoresf, CT, CS, CD, CD, CS, CS,
         (long)CH * CT * CD, (long)CT * CD, (long)CD * CS, 0,
         (long)CH * CT * CS, (long)CT * CS, CH, CB * CH, 0);
    softmax_kernel<<<dim3((unsigned)(CB * CH * CT / 8)), 256, 0, stream>>>(scoresf, nullptr, cbiasf, attn_h, CS);
    // cctx = cattn @ cv : per-(b,h) (512 x 64 x 512) -> (B,T,H,DH) in vf
    gemm(stream, attn_h, v_h, nullptr, vf, CT, CDH, CS, CS, CD, CD,
         (long)CH * CT * CS, (long)CT * CS, (long)CS * CD, CDH,
         (long)CT * CD, CDH, CH, CB * CH, 0);
    cvt(stream, vf, ctx_h, NN, 1.f);
    gemm(stream, ctx_h, coT, ca_out_b, projf, (int)MROWS, CD, CD, CD, CD, CD, 0,0,0,0,0,0, 1, 1, 0);
    residual_ln_kernel<<<dim3((unsigned)(MROWS / 8)), 256, 0, stream>>>(x1f, projf, ln2_g, ln2_b, x2f, x2h);

    // ---- FFN ----
    gemm(stream, x2h, f1T, ff1_b, h1f, (int)MROWS, 4 * CD, CD, CD, 4 * CD, 4 * CD, 0,0,0,0,0,0, 1, 1, 1);
    cvt(stream, h1f, h1h, NFF, 1.f);
    gemm(stream, h1h, f2T, ff2_b, projf, (int)MROWS, CD, 4 * CD, 4 * CD, CD, CD, 0,0,0,0,0,0, 1, 1, 0);
    residual_ln_kernel<<<dim3((unsigned)(MROWS / 8)), 256, 0, stream>>>(x2f, projf, ln3_g, ln3_b, outp, nullptr);
}
